// MultiHeadAttention_24713241821329
// MI455X (gfx1250) — matrix operand order repaired and checked by reading, compiled, not run on hardware
//
#include <hip/hip_runtime.h>
#include <hip/hip_bf16.h>

// Problem constants (from reference)
#define B_  2
#define S_  2048
#define H_  1024
#define NH_ 16
#define HD_ 64
#define BS_ (B_*S_)   // 4096

typedef __attribute__((ext_vector_type(16))) __bf16 v16bf;
typedef __attribute__((ext_vector_type(8)))  __bf16 v8bf;
typedef __attribute__((ext_vector_type(8)))  float  v8f;

union Frag16 { v16bf v; v8bf h[2]; };

__device__ __forceinline__ int lane_id() { return (int)(threadIdx.x & 31u); }

// A operand (16x32 bf16, row-major source with leading dim ld).
// ISA layout: lanes 0-15 = rows 0-15; element i -> K = (i>>3)*16 + (lane>=16?8:0) + (i&7)
__device__ __forceinline__ v16bf load_a_frag(const __bf16* p, int ld, int row0, int k0) {
  int lane = lane_id();
  int m = lane & 15, hi = lane >> 4;
  const __bf16* base = p + (size_t)(row0 + m) * ld + (k0 + hi * 8);
  Frag16 f;
  f.h[0] = *(const v8bf*)(base);
  f.h[1] = *(const v8bf*)(base + 16);
  return f.v;
}

// B operand (32x16 bf16). Source stored [col][k] (k contiguous), leading dim ld.
// ISA layout: lane = column (n = lane&15); element i -> K = (i>>3)*16 + (lane>=16?8:0) + (i&7)
__device__ __forceinline__ v16bf load_b_frag(const __bf16* p, int ld, int col0, int k0) {
  int lane = lane_id();
  int n = lane & 15, hi = lane >> 4;
  const __bf16* base = p + (size_t)(col0 + n) * ld + (k0 + hi * 8);
  Frag16 f;
  f.h[0] = *(const v8bf*)(base);
  f.h[1] = *(const v8bf*)(base + 16);
  return f.v;
}

__device__ __forceinline__ v8f wmma_bf16(v16bf a, v16bf b, v8f c) {
  // D = A*B + C, fp32 accumulate
  return __builtin_amdgcn_wmma_f32_16x16x32_bf16(false, a, false, b, (short)0, c, false, false);
}

// ---------------------------------------------------------------------------
__global__ void cast_f32_to_bf16_kernel(const float* __restrict__ in,
                                        __bf16* __restrict__ out, int n) {
  int i = blockIdx.x * blockDim.x + threadIdx.x;
  if (i < n) out[i] = (__bf16)in[i];
}

// W[k][n] (f32) -> WT[n][k] (bf16)
__global__ void transpose_cast_kernel(const float* __restrict__ W,
                                      __bf16* __restrict__ WT, int K, int N) {
  int i = blockIdx.x * blockDim.x + threadIdx.x;
  if (i < K * N) {
    int k = i / N, n = i - k * N;
    WT[(size_t)n * K + k] = (__bf16)W[i];
  }
}

// C[M][N] = A[M][K] @ WT[N][K]^T + bias.
// 4 waves/block; each wave computes a 32x64 tile (8 accumulators).
// K-loop unrolled x2 with explicit ping-pong fragment buffers.
template <bool F32OUT>
__global__ void __launch_bounds__(128, 1)
gemm_bias_kernel(const __bf16* __restrict__ A,
                 const __bf16* __restrict__ WT,
                 const float* __restrict__ bias,
                 void* __restrict__ outp,
                 int M, int N, int K) {
  int wave = threadIdx.x >> 5;
  int lane = lane_id();
  int n0 = blockIdx.x * 64;
  int m0 = blockIdx.y * 128 + wave * 32;

  v8f acc[8] = {};
  auto mma8 = [&](const v16bf (&aa)[2], const v16bf (&bb)[4]) {
#pragma unroll
    for (int mt = 0; mt < 2; ++mt)
#pragma unroll
      for (int nt = 0; nt < 4; ++nt)
        acc[mt * 4 + nt] = wmma_bf16(aa[mt], bb[nt], acc[mt * 4 + nt]);
  };

  v16bf a0[2], b0[4], a1[2], b1[4];
  a0[0] = load_a_frag(A, K, m0,      0);
  a0[1] = load_a_frag(A, K, m0 + 16, 0);
#pragma unroll
  for (int nt = 0; nt < 4; ++nt) b0[nt] = load_b_frag(WT, K, n0 + nt * 16, 0);

  for (int k0 = 0; k0 < K; k0 += 64) {
    // fill buf1 with slab k0+32 (always in range: K is a multiple of 64)
    int k1 = k0 + 32;
    a1[0] = load_a_frag(A, K, m0,      k1);
    a1[1] = load_a_frag(A, K, m0 + 16, k1);
#pragma unroll
    for (int nt = 0; nt < 4; ++nt) b1[nt] = load_b_frag(WT, K, n0 + nt * 16, k1);

    mma8(a0, b0);   // consume buf0 while buf1 is in flight

    // fill buf0 with slab k0+64 (wrap on last iteration; harmless reload)
    int k2 = (k0 + 64 < K) ? (k0 + 64) : 0;
    __builtin_prefetch(A + (size_t)(m0 + (lane & 15)) * K + k2, 0, 3);
    a0[0] = load_a_frag(A, K, m0,      k2);
    a0[1] = load_a_frag(A, K, m0 + 16, k2);
#pragma unroll
    for (int nt = 0; nt < 4; ++nt) b0[nt] = load_b_frag(WT, K, n0 + nt * 16, k2);

    mma8(a1, b1);   // consume buf1 while buf0 is in flight
  }

  int hi = lane >> 4, n = lane & 15;
  float* of = (float*)outp;
  __bf16* ob = (__bf16*)outp;
#pragma unroll
  for (int mt = 0; mt < 2; ++mt)
#pragma unroll
    for (int nt = 0; nt < 4; ++nt) {
      int col = n0 + nt * 16 + n;
      float bv = bias[col];
#pragma unroll
      for (int r = 0; r < 8; ++r) {
        int row = m0 + mt * 16 + r + hi * 8;
        float val = acc[mt * 4 + nt][r] + bv;
        if (F32OUT) of[(size_t)row * N + col] = val;
        else        ob[(size_t)row * N + col] = (__bf16)val;
      }
    }
}

// In-place RoPE on Q and K. One thread per (b,s,h,i<32) pair (d=i and d=i+32 share angle).
__global__ void rope_kernel(__bf16* __restrict__ Q, __bf16* __restrict__ Km) {
  int idx = blockIdx.x * blockDim.x + threadIdx.x;
  if (idx >= B_ * S_ * NH_ * 32) return;
  int i  = idx & 31;
  int t  = idx >> 5;            // (b*S + s)*NH + h
  int h  = t & (NH_ - 1);
  int bs = t >> 4;              // b*S + s
  int s  = bs & (S_ - 1);
  size_t base = (size_t)bs * H_ + h * HD_;
  float ang = (float)s * __powf(10000.0f, -(float)i * (1.0f / 32.0f));
  float c = __cosf(ang), sn = __sinf(ang);
  float q1 = (float)Q[base + i], q2 = (float)Q[base + i + 32];
  Q[base + i]       = (__bf16)(q1 * c - q2 * sn);
  Q[base + i + 32]  = (__bf16)(q2 * c + q1 * sn);
  float k1 = (float)Km[base + i], k2 = (float)Km[base + i + 32];
  Km[base + i]      = (__bf16)(k1 * c - k2 * sn);
  Km[base + i + 32] = (__bf16)(k2 * c + k1 * sn);
}

// V [b,s,h*64+d] -> VT [b,h,d,s]  (so P@V B-operand is key-contiguous)
__global__ void transpose_v_kernel(const __bf16* __restrict__ V,
                                   __bf16* __restrict__ VT) {
  int idx = blockIdx.x * blockDim.x + threadIdx.x;   // over B*S*H
  if (idx >= B_ * S_ * H_) return;
  int c  = idx & (H_ - 1);
  int bs = idx >> 10;
  int b  = bs >> 11;
  int s  = bs & (S_ - 1);
  int h  = c >> 6, d = c & 63;
  VT[(((size_t)b * NH_ + h) * HD_ + d) * S_ + s] = V[idx];
}

// One flash-attention key block (32 keys). MASKED only for the single
// diagonal block (kb == q0>>5); all earlier blocks are entirely causal-legal,
// so the main loop carries no compares / row-index registers at all.
template <bool MASKED>
__device__ __forceinline__ void fa_block(
    int kb, int q0, int hi, int n,
    const __bf16* __restrict__ Kp, const __bf16* __restrict__ Vp,
    __bf16 (*Pw)[32],               // this wave's P staging tile [16][32] in LDS
    v16bf qf0, v16bf qf1,
    v8f& o0, v8f& o1, v8f& o2, v8f& o3,
    float (&run_max)[8], float (&run_sum)[8]) {
  v8f s0 = {}, s1 = {};
#pragma unroll
  for (int db = 0; db < 2; ++db) {          // head-dim split 64 = 2 x 32
    v16bf qf  = db ? qf1 : qf0;
    v16bf kf0 = load_b_frag(Kp, H_, kb * 32,      db * 32);
    v16bf kf1 = load_b_frag(Kp, H_, kb * 32 + 16, db * 32);
    s0 = wmma_bf16(qf, kf0, s0);
    s1 = wmma_bf16(qf, kf1, s1);
  }
  // V fragments issued early: independent of the softmax below
  v16bf vf0 = load_b_frag(Vp, S_, 0,  kb * 32);
  v16bf vf1 = load_b_frag(Vp, S_, 16, kb * 32);
  v16bf vf2 = load_b_frag(Vp, S_, 32, kb * 32);
  v16bf vf3 = load_b_frag(Vp, S_, 48, kb * 32);

  // scale (1/sqrt(64)) [+ causal mask on the diagonal block] + per-lane row max
  float nmax[8];
#pragma unroll
  for (int r = 0; r < 8; ++r) {
    float a0 = s0[r] * 0.125f;
    float a1 = s1[r] * 0.125f;
    if (MASKED) {
      int row  = q0 + r + hi * 8;
      int key0 = kb * 32 + n;
      if (key0 > row)      a0 = -3.0e38f;
      if (key0 + 16 > row) a1 = -3.0e38f;
    }
    s0[r] = a0; s1[r] = a1;
    nmax[r] = fmaxf(run_max[r], fmaxf(a0, a1));
  }
#pragma unroll
  for (int mask = 1; mask < 16; mask <<= 1)
#pragma unroll
    for (int r = 0; r < 8; ++r)
      nmax[r] = fmaxf(nmax[r], __shfl_xor(nmax[r], mask, 32));
  float psum[8];
#pragma unroll
  for (int r = 0; r < 8; ++r) {
    float p0 = __expf(s0[r] - nmax[r]);
    float p1 = __expf(s1[r] - nmax[r]);
    s0[r] = p0; s1[r] = p1;
    psum[r] = p0 + p1;
  }
#pragma unroll
  for (int mask = 1; mask < 16; mask <<= 1)
#pragma unroll
    for (int r = 0; r < 8; ++r)
      psum[r] += __shfl_xor(psum[r], mask, 32);
#pragma unroll
  for (int r = 0; r < 8; ++r) {
    float alpha = __expf(run_max[r] - nmax[r]);
    run_sum[r] = run_sum[r] * alpha + psum[r];
    run_max[r] = nmax[r];
    o0[r] *= alpha; o1[r] *= alpha; o2[r] *= alpha; o3[r] *= alpha;
  }
  // D-layout -> LDS -> A-fragment (same-wave DS ops are in order; waits inserted)
#pragma unroll
  for (int r = 0; r < 8; ++r) {
    Pw[r + hi * 8][n]      = (__bf16)s0[r];
    Pw[r + hi * 8][16 + n] = (__bf16)s1[r];
  }
  v16bf pf = load_a_frag(&Pw[0][0], 32, 0, 0);
  o0 = wmma_bf16(pf, vf0, o0);
  o1 = wmma_bf16(pf, vf1, o1);
  o2 = wmma_bf16(pf, vf2, o2);
  o3 = wmma_bf16(pf, vf3, o3);
}

// Flash attention: 1 wave per 16-query tile, key blocks of 32, online softmax.
__global__ void __launch_bounds__(128, 1)
flash_attn_kernel(const __bf16* __restrict__ Q,
                  const __bf16* __restrict__ Km,
                  const __bf16* __restrict__ VT,
                  __bf16* __restrict__ O) {
  __shared__ __bf16 Pl[4][16][32];  // per-wave P staging (D-layout -> A-layout)
  int wave = threadIdx.x >> 5;
  int lane = lane_id();
  int hi = lane >> 4, n = lane & 15;

  int qc = blockIdx.x & 31;         // S/64 = 32 query chunks
  int bh = blockIdx.x >> 5;
  int h  = bh & (NH_ - 1);
  int b  = bh >> 4;
  int q0 = qc * 64 + wave * 16;

  const __bf16* Qp = Q  + (size_t)b * S_ * H_ + h * HD_;
  const __bf16* Kp = Km + (size_t)b * S_ * H_ + h * HD_;
  const __bf16* Vp = VT + ((size_t)b * NH_ + h) * (size_t)HD_ * S_;

  v16bf qf0 = load_a_frag(Qp, H_, q0, 0);
  v16bf qf1 = load_a_frag(Qp, H_, q0, 32);

  v8f o0 = {}, o1 = {}, o2 = {}, o3 = {};
  float run_max[8], run_sum[8];
#pragma unroll
  for (int r = 0; r < 8; ++r) { run_max[r] = -3.0e38f; run_sum[r] = 0.0f; }

  // Blocks [0, q0>>5) are fully below the diagonal: mask-free main loop.
  int nfull = q0 >> 5;
  for (int kb = 0; kb < nfull; ++kb)
    fa_block<false>(kb, q0, hi, n, Kp, Vp, Pl[wave], qf0, qf1,
                    o0, o1, o2, o3, run_max, run_sum);
  // Single partially-masked diagonal block (always exists; q0 % 16 == 0).
  fa_block<true>(nfull, q0, hi, n, Kp, Vp, Pl[wave], qf0, qf1,
                 o0, o1, o2, o3, run_max, run_sum);

  // normalize + store [b, q, h*64 + d]
#pragma unroll
  for (int r = 0; r < 8; ++r) {
    float inv = 1.0f / run_sum[r];
    int row = q0 + r + hi * 8;
    size_t base = ((size_t)b * S_ + row) * H_ + h * HD_ + n;
    O[base]      = (__bf16)(o0[r] * inv);
    O[base + 16] = (__bf16)(o1[r] * inv);
    O[base + 32] = (__bf16)(o2[r] * inv);
    O[base + 48] = (__bf16)(o3[r] * inv);
  }
}

// ---------------------------------------------------------------------------
extern "C" void kernel_launch(void* const* d_in, const int* in_sizes, int n_in,
                              void* d_out, int out_size, void* d_ws, size_t ws_size,
                              hipStream_t stream) {
  const float* hidden = (const float*)d_in[0];
  // d_in[1] = attention_mask: causal, reproduced analytically in-kernel.
  const float* q_w = (const float*)d_in[2];
  const float* q_b = (const float*)d_in[3];
  const float* k_w = (const float*)d_in[4];
  const float* k_b = (const float*)d_in[5];
  const float* v_w = (const float*)d_in[6];
  const float* v_b = (const float*)d_in[7];
  const float* o_w = (const float*)d_in[8];
  const float* o_b = (const float*)d_in[9];
  float* out = (float*)d_out;

  char* ws = (char*)d_ws;
  const size_t MB = 1024u * 1024u;
  __bf16* hid_bf = (__bf16*)(ws);            //  8 MiB
  __bf16* wqT    = (__bf16*)(ws +  8 * MB);  //  2 MiB each
  __bf16* wkT    = (__bf16*)(ws + 10 * MB);
  __bf16* wvT    = (__bf16*)(ws + 12 * MB);
  __bf16* woT    = (__bf16*)(ws + 14 * MB);
  __bf16* qbuf   = (__bf16*)(ws + 16 * MB);  //  8 MiB each
  __bf16* kbuf   = (__bf16*)(ws + 24 * MB);
  __bf16* vbuf   = (__bf16*)(ws + 32 * MB);
  __bf16* vT     = (__bf16*)(ws + 40 * MB);
  __bf16* obuf   = (__bf16*)(ws + 48 * MB);  // total 56 MiB

  const int nh = BS_ * H_;                   // 4,194,304
  cast_f32_to_bf16_kernel<<<(nh + 255) / 256, 256, 0, stream>>>(hidden, hid_bf, nh);

  const int nw = H_ * H_;
  transpose_cast_kernel<<<(nw + 255) / 256, 256, 0, stream>>>(q_w, wqT, H_, H_);
  transpose_cast_kernel<<<(nw + 255) / 256, 256, 0, stream>>>(k_w, wkT, H_, H_);
  transpose_cast_kernel<<<(nw + 255) / 256, 256, 0, stream>>>(v_w, wvT, H_, H_);
  transpose_cast_kernel<<<(nw + 255) / 256, 256, 0, stream>>>(o_w, woT, H_, H_);

  dim3 gg(H_ / 64, BS_ / 128), bbk(128);     // wave tile 32x64, block tile 128x64
  gemm_bias_kernel<false><<<gg, bbk, 0, stream>>>(hid_bf, wqT, q_b, qbuf, BS_, H_, H_);
  gemm_bias_kernel<false><<<gg, bbk, 0, stream>>>(hid_bf, wkT, k_b, kbuf, BS_, H_, H_);
  gemm_bias_kernel<false><<<gg, bbk, 0, stream>>>(hid_bf, wvT, v_b, vbuf, BS_, H_, H_);

  const int nr = B_ * S_ * NH_ * 32;
  rope_kernel<<<(nr + 255) / 256, 256, 0, stream>>>(qbuf, kbuf);
  transpose_v_kernel<<<(nh + 255) / 256, 256, 0, stream>>>(vbuf, vT);

  flash_attn_kernel<<<B_ * NH_ * (S_ / 64), 128, 0, stream>>>(qbuf, kbuf, vT, obuf);

  gemm_bias_kernel<true><<<gg, bbk, 0, stream>>>(obuf, woT, o_b, out, BS_, H_, H_);

  (void)in_sizes; (void)n_in; (void)out_size; (void)ws_size;
}
